// SAGE_TEST_63814624084655
// MI455X (gfx1250) — matrix operand order repaired and checked by reading, not compiled, not run on hardware
//
#include <hip/hip_runtime.h>

typedef __attribute__((ext_vector_type(16))) _Float16 v16h;
typedef __attribute__((ext_vector_type(8)))  float    v8f;

#define Bsz 128
#define Nn  512
#define Kn  10
#define Ff  128
#define CF  256   // 2*F
#define Cc  40

union F16x16 { v16h v; _Float16 e[16]; };

// Convert both SAGE layers' weights (L,F,2F) fp32 -> f16, row-major.
// 2*128*256 = 65536 elements.
__global__ __launch_bounds__(256) void prep_w16_kernel(
    const float* __restrict__ w, _Float16* __restrict__ w16)
{
    int i = blockIdx.x * 256 + threadIdx.x;   // grid covers 65536
    w16[i] = (_Float16)w[i];
}

// One workgroup computes a 16-row x 128-col output tile of one SAGE layer.
// layer2 != 0: rows map to (b = tile*16 + m, n = 0)  [only h[:,0,:] is needed]
// layer2 == 0: rows map to (b = tile>>5, n = (tile&31)*16 + m)
__global__ __launch_bounds__(256) void sage_wmma_kernel(
    const float* __restrict__ hin, const int* __restrict__ neigh,
    const _Float16* __restrict__ w16, const float* __restrict__ bias,
    float* __restrict__ hout, int layer2)
{
    __shared__ _Float16 cat[16][264];   // 16 rows x 256 (mp | h), padded
    __shared__ int      nbs[16][Kn];    // staged neighbor indices
    const int tid  = threadIdx.x;
    const int tile = blockIdx.x;

    // ---- stage neighbor indices for the 16 rows of this tile ----
    if (tid < 16 * Kn) {
        int m = tid / Kn, k = tid % Kn;
        int b, n;
        if (layer2) { b = tile * 16 + m; n = 0; }
        else        { b = tile >> 5;     n = ((tile & 31) << 4) + m; }
        nbs[m][k] = neigh[((size_t)b * Nn + n) * Kn + k];
    }
    __syncthreads();

    // ---- cooperative mean-pool + concat into LDS as f16 ----
    #pragma unroll
    for (int i = 0; i < 8; ++i) {
        int e = tid + i * 256;              // 0..2047 covers 16x128
        int m = e >> 7;
        int f = e & 127;
        int b, n;
        if (layer2) { b = tile * 16 + m; n = 0; }
        else        { b = tile >> 5;     n = ((tile & 31) << 4) + m; }
        const float* hb = hin + (size_t)b * Nn * Ff;
        float s = 0.f;
        #pragma unroll
        for (int k = 0; k < Kn; ++k)
            s += hb[(size_t)nbs[m][k] * Ff + f];
        cat[m][f]      = (_Float16)(s * 0.1f);          // mean over K=10
        cat[m][Ff + f] = (_Float16)hb[(size_t)n * Ff + f]; // self features
    }
    __syncthreads();

    // ---- WMMA: out(16x128) = cat(16x256) x W^T(256x128), fp32 accum ----
    const int lane = tid & 31;
    const int wave = tid >> 5;
    const int g0   = wave << 4;     // each wave owns 16 output columns
    const int nsel = lane & 15;
    const int grp  = lane >> 4;     // 0 = lanes 0-15, 1 = lanes 16-31
    const int g    = g0 + nsel;

    v8f c = {};
    #pragma unroll
    for (int ks = 0; ks < 8; ++ks) {
        // B fragment (32x16): lane holds col nsel, k_local = grp*16 + hi.
        // B[k][n] = w[g0+n][ks*32 + k]; per-lane 16 halves are contiguous
        // and 32B-aligned -> single vector load.
        F16x16 bf;
        bf.v = *(const v16h*)(w16 + (size_t)g * CF + ks * 32 + grp * 16);

        // A fragment (16x32): lane holds row m = nsel
        // grp=0: hi 0-15 -> K 0-15
        // grp=1: hi 0-15 -> K 16-31
        F16x16 af;
        #pragma unroll
        for (int hi = 0; hi < 8; ++hi) {
            af.e[hi]     = cat[nsel][ks * 32 + grp * 16 + hi];
            af.e[hi + 8] = cat[nsel][ks * 32 + grp * 16 + 8 + hi];
        }

        c = __builtin_amdgcn_wmma_f32_16x16x32_f16(
                /*neg_a=*/false, af.v, /*neg_b=*/false, bf.v,
                /*c_mod=*/(short)0, c, /*reuse_a=*/false, /*reuse_b=*/false);
    }

    // C/D layout: VGPR r -> M = r + grp*8, N = nsel
    const float bv = bias[g];
    #pragma unroll
    for (int r = 0; r < 8; ++r) {
        int   M   = r + grp * 8;
        float val = c[r] + bv;
        if (layer2) {
            hout[(size_t)(tile * 16 + M) * Ff + g] = val;
        } else {
            int b = tile >> 5;
            int n = ((tile & 31) << 4) + M;
            hout[((size_t)b * Nn + n) * Ff + g] = val;
        }
    }
}

// Single-block head: lin1 + batchnorm(batch) + relu + lin2 + softmax.
__global__ __launch_bounds__(256) void head_kernel(
    const float* __restrict__ feat,   // 128 x 128
    const float* __restrict__ l1w, const float* __restrict__ l1b,
    const float* __restrict__ gam, const float* __restrict__ bet,
    const float* __restrict__ l2w, const float* __restrict__ l2b,
    float* __restrict__ out)          // 128 x 40 (softmax probs)
{
    __shared__ float z[128][64];
    __shared__ float mu[64], rstd[64];
    __shared__ float act[128][40];
    const int tid = threadIdx.x;

    // z = feat @ lin1_w^T + lin1_b
    for (int i = 0; i < 32; ++i) {
        int e = tid + i * 256;          // 8192 = 128*64
        int row = e >> 6, j = e & 63;
        float s = l1b[j];
        for (int f = 0; f < 128; ++f)
            s += feat[row * 128 + f] * l1w[j * 128 + f];
        z[row][j] = s;
    }
    __syncthreads();

    // batchnorm stats over the 128-row batch
    if (tid < 64) {
        float s = 0.f;
        for (int r = 0; r < 128; ++r) s += z[r][tid];
        float m = s * (1.f / 128.f);
        float v = 0.f;
        for (int r = 0; r < 128; ++r) { float d = z[r][tid] - m; v += d * d; }
        v *= (1.f / 128.f);
        mu[tid]   = m;
        rstd[tid] = rsqrtf(v + 1e-5f);
    }
    __syncthreads();

    // normalize + affine + relu
    for (int i = 0; i < 32; ++i) {
        int e = tid + i * 256;
        int row = e >> 6, j = e & 63;
        float t = (z[row][j] - mu[j]) * rstd[j] * gam[j] + bet[j];
        z[row][j] = t > 0.f ? t : 0.f;
    }
    __syncthreads();

    // logits = z @ lin2_w^T + lin2_b
    for (int i = 0; i < 20; ++i) {
        int e = tid + i * 256;          // 5120 = 128*40
        int row = e / 40, cc = e % 40;
        float s = l2b[cc];
        for (int k = 0; k < 64; ++k)
            s += z[row][k] * l2w[cc * 64 + k];
        act[row][cc] = s;
    }
    __syncthreads();

    // row softmax
    if (tid < 128) {
        float mx = -1e30f;
        for (int cc = 0; cc < Cc; ++cc) mx = fmaxf(mx, act[tid][cc]);
        float s = 0.f;
        for (int cc = 0; cc < Cc; ++cc) s += expf(act[tid][cc] - mx);
        float inv = 1.f / s;
        for (int cc = 0; cc < Cc; ++cc)
            out[tid * Cc + cc] = expf(act[tid][cc] - mx) * inv;
    }
}

extern "C" void kernel_launch(void* const* d_in, const int* in_sizes, int n_in,
                              void* d_out, int out_size, void* d_ws, size_t ws_size,
                              hipStream_t stream) {
    const float* x     = (const float*)d_in[0];
    const int*   neigh = (const int*)  d_in[1];
    const float* wsage = (const float*)d_in[2];   // (L, F, 2F)
    const float* bsage = (const float*)d_in[3];   // (L, F)
    const float* l1w   = (const float*)d_in[4];
    const float* l1b   = (const float*)d_in[5];
    const float* gam   = (const float*)d_in[6];
    const float* bet   = (const float*)d_in[7];
    const float* l2w   = (const float*)d_in[8];
    const float* l2b   = (const float*)d_in[9];
    float* out = (float*)d_out;

    // Workspace layout (all offsets 32B-aligned):
    float*     h1  = (float*)d_ws;                        // B*N*F fp32 (~33.5 MB)
    float*     h2  = h1 + (size_t)Bsz * Nn * Ff;          // B*F fp32 (64 KB)
    _Float16*  w16 = (_Float16*)(h2 + (size_t)Bsz * Ff);  // L*F*2F f16 (128 KB)

    // Convert SAGE weights to f16 once (2*128*256 = 65536 elems).
    prep_w16_kernel<<<(2 * Ff * CF) / 256, 256, 0, stream>>>(wsage, w16);
    // Layer 0: full graph, 4096 tiles of 16 rows.
    sage_wmma_kernel<<<Bsz * (Nn / 16), 256, 0, stream>>>(
        x, neigh, w16, bsage, h1, 0);
    // Layer 1: only n=0 rows are consumed downstream -> 128 rows, 8 tiles.
    sage_wmma_kernel<<<Bsz / 16, 256, 0, stream>>>(
        h1, neigh, w16 + (size_t)Ff * CF, bsage + Ff, h2, 1);
    // Head: single block.
    head_kernel<<<1, 256, 0, stream>>>(h2, l1w, l1b, gam, bet, l2w, l2b, out);
}
